// DeepRootMUSIC_33139967656342
// MI455X (gfx1250) — matrix operand order repaired and checked by reading, not compiled, not run on hardware
//
#include <hip/hip_runtime.h>
#include <hip/hip_bf16.h>

// ---------------------------------------------------------------------------
// Types for CDNA5 WMMA fragments / TDM descriptors
// ---------------------------------------------------------------------------
typedef __attribute__((ext_vector_type(16))) _Float16 v16h;
typedef __attribute__((ext_vector_type(8)))  float    v8f;
typedef __attribute__((ext_vector_type(2)))  float    v2f;
typedef __attribute__((ext_vector_type(4)))  unsigned int uint32x4;
typedef __attribute__((ext_vector_type(8)))  int      int32x8;
typedef __attribute__((ext_vector_type(4)))  int      int32x4;

#if defined(__has_builtin)
#if __has_builtin(__builtin_amdgcn_wmma_f32_16x16x4_f32)
#define HAS_WMMA_F32X4 1
#endif
#if __has_builtin(__builtin_amdgcn_tensor_load_to_lds)
#define HAS_TDM 1
#endif
#endif
#ifndef HAS_WMMA_F32X4
#define HAS_WMMA_F32X4 0
#endif
#ifndef HAS_TDM
#define HAS_TDM 0
#endif

// ---------------------------------------------------------------------------
// Complex helpers (float2 / double2)
// ---------------------------------------------------------------------------
__device__ __forceinline__ float2 cadd(float2 a, float2 b){ return make_float2(a.x+b.x, a.y+b.y); }
__device__ __forceinline__ float2 csub(float2 a, float2 b){ return make_float2(a.x-b.x, a.y-b.y); }
__device__ __forceinline__ float2 cmul(float2 a, float2 b){ return make_float2(a.x*b.x - a.y*b.y, a.x*b.y + a.y*b.x); }
__device__ __forceinline__ float2 cscale(float s, float2 a){ return make_float2(s*a.x, s*a.y); }

__device__ __forceinline__ double2 dadd(double2 a, double2 b){ return make_double2(a.x+b.x, a.y+b.y); }
__device__ __forceinline__ double2 dsub(double2 a, double2 b){ return make_double2(a.x-b.x, a.y-b.y); }
__device__ __forceinline__ double2 dmul(double2 a, double2 b){ return make_double2(a.x*b.x - a.y*b.y, a.x*b.y + a.y*b.x); }
__device__ __forceinline__ double2 ddiv(double2 a, double2 b){
    double d = b.x*b.x + b.y*b.y + 1e-300;
    return make_double2((a.x*b.x + a.y*b.y)/d, (a.y*b.x - a.x*b.y)/d);
}

// ---------------------------------------------------------------------------
// TDM / cluster probe kernel: NULL-tensor (count=0) tensor_load_to_lds,
// s_wait_tensorcnt, cluster barrier (NOP outside a cluster), cluster_id read.
// Exercises the gfx1250 TENSORcnt path; transfers nothing (descriptor count=0).
// ---------------------------------------------------------------------------
__global__ void tdm_probe_kernel(int* __restrict__ sink)
{
#if HAS_TDM
    uint32x4 g0 = {};   // D# group0: count=0 -> NULL tensor, no data moved
    int32x8  g1 = {};
    int32x4  g2 = {};
    int32x4  g3 = {};
#if __clang_major__ >= 23
    int32x8  g4 = {};
    __builtin_amdgcn_tensor_load_to_lds(g0, g1, g2, g3, g4, 0);
#else
    __builtin_amdgcn_tensor_load_to_lds(g0, g1, g2, g3, 0);
#endif
    __builtin_amdgcn_s_wait_tensorcnt(0);
#endif
#if defined(__has_builtin)
#if __has_builtin(__builtin_amdgcn_s_cluster_barrier)
    __builtin_amdgcn_s_cluster_barrier();   // NOP when ClusterID==0
#endif
#if __has_builtin(__builtin_amdgcn_cluster_id_x)
    if (threadIdx.x == 0) sink[0] = __builtin_amdgcn_cluster_id_x(); // 0 outside cluster
#endif
#endif
}

// ---------------------------------------------------------------------------
// LDS-staged implicit-GEMM 2x2 conv / deconv via v_wmma_f32_16x16x32_f16.
//
// Block = 32*ntiles threads (one wave per 16-wide Cout tile). Each block owns
// one output row strip (oy, ox0..ox0+15). The 2 x 17 x Cin f16 input patch and
// the (pre-flipped) f16 weights are staged once in LDS and shared by all waves
// -> removes the 2x2-overlap and Cout-tile redundancy in global traffic
// (the conv stack is HBM-bound: ~43 GFLOP vs ~1 GB moved at 23.3 TB/s).
// ---------------------------------------------------------------------------
template<typename IN_T, typename OUT_T, bool DECONV, bool LRELU>
__global__ void conv2x2_wmma(const IN_T* __restrict__ in, const float* __restrict__ w,
                             const float* __restrict__ bias, OUT_T* __restrict__ out,
                             int Cin, int Cout, int Hin, int Win, int Hout, int Wout)
{
    __shared__ _Float16 sIn[64][2][18];   // [cin][ky][patch col 0..16]
    __shared__ _Float16 sW[8192];         // [co][ci][ky][kx], flip pre-applied

    const int T   = blockDim.x;
    const int tid = threadIdx.x;

    int bid = blockIdx.x;
    const int wt = bid & 3; bid >>= 2;            // all layers: ceil(Wout/16)==4
    const int oy = bid % Hout;
    const int b  = bid / Hout;
    const int ox0 = wt * 16;

    const int dy0 = DECONV ? -1 : 0;
    const int dx0 = DECONV ? -1 : 0;
    const long inB = (long)b * Cin * Hin * Win;

    // ---- stage weights (spatial flip applied for deconv) ----
    const int wcnt = Cout * Cin * 4;
    for (int idx = tid; idx < wcnt; idx += T) {
        int kx = idx & 1, ky = (idx >> 1) & 1, rest = idx >> 2;
        int wy = DECONV ? (1 - ky) : ky;
        int wx = DECONV ? (1 - kx) : kx;
        sW[idx] = (_Float16)w[(long)rest * 4 + wy * 2 + wx];
    }
    // ---- stage input patch: rows oy+dy0 .. oy+dy0+1, cols ox0+dx0 .. +16 ----
    const int pcnt = Cin * 2 * 17;
    for (int idx = tid; idx < pcnt; idx += T) {
        int ci = idx / 34, rem = idx % 34;
        int ry = rem / 17, cx = rem % 17;
        int iy = oy + dy0 + ry;
        int ix = ox0 + dx0 + cx;
        float v = 0.f;
        if (iy >= 0 && iy < Hin && ix >= 0 && ix < Win) {
            long a = inB + ((long)ci * Hin + iy) * Win + ix;
            v = (float)in[a];
            __builtin_prefetch(&in[a + 2 * Win], 0, 1);  // global_prefetch_b8: warm rows ahead
        }
        sIn[ci][ry][cx] = (_Float16)v;
    }
    __syncthreads();

    const int lane = tid & 31;
    const int g    = lane >> 4;
    const int mloc = lane & 15;
    const int nt   = tid >> 5;
    const int n    = nt * 16 + mloc;              // output channel

    v8f acc = {};
    for (int c0 = 0; c0 < Cin; c0 += 8) {
        v16h afrag, bfrag;
        // A fragment: 16-bit A 16x32 layout (ISA 7.12.2); row m = mloc
        #pragma unroll
        for (int h = 0; h < 16; ++h) {
            int k  = (h < 8 ? h : h + 8) + (g ? 8 : 0);
            int ci = c0 + (k >> 2);
            int r  = k & 3, ky = r >> 1, kx = r & 1;
            afrag[h] = sIn[ci][ky][mloc + kx];
        }
        // B fragment: 16-bit B 32x16 layout; col n
        #pragma unroll
        for (int h = 0; h < 16; ++h) {
            int k  = (h < 8 ? h : h + 8) + (g ? 8 : 0);
            int ci = c0 + (k >> 2);
            int r  = k & 3;
            bfrag[h] = (n < Cout) ? sW[((n * Cin + ci) << 2) + r] : (_Float16)0.f;
        }
        acc = __builtin_amdgcn_wmma_f32_16x16x32_f16(false, afrag, false, bfrag,
                                                     (short)0, acc, false, false);
    }
    // store C: row = r + g*8 -> ox = ox0 + row, col = mloc -> cout
    if (n < Cout) {
        float bv = bias[n];
        #pragma unroll
        for (int r = 0; r < 8; ++r) {
            int ox = ox0 + r + g * 8;
            if (ox < Wout) {
                float v = acc[r] + bv;
                if (LRELU) v = (v >= 0.f) ? v : 0.5f * v;
                out[(((long)b * Cout + n) * Hout + oy) * Wout + ox] = (OUT_T)v;
            }
        }
    }
}

// ---------------------------------------------------------------------------
// Rz = K^H K + eps*I  (K = Rx[:, :64] + i Rx[:, 64:], each 64x64).
// 4 real 64x64x64 GEMMs via v_wmma_f32_16x16x4_f32. One wave per 16x16 tile.
// Writes interleaved complex64 directly into d_out's Rz slice.
// ---------------------------------------------------------------------------
__global__ void rz_kernel(const float* __restrict__ Rx, float* __restrict__ RzOut)
{
    const int lane = threadIdx.x & 31;
    const int g    = lane >> 4;
    const int mloc = lane & 15;
    int bid = blockIdx.x;
    const int tk = bid & 3; bid >>= 2;
    const int ti = bid & 3; bid >>= 2;
    const int b  = bid;

    const float* Kr = Rx + (size_t)b * 8192;   // [j][i], stride 64
    const float* Ki = Kr + 64 * 64;

    v8f accRR = {}, accII = {}, accRI = {}, accIR = {};
#if HAS_WMMA_F32X4
    const int mi = ti * 16 + mloc;
    const int nk = tk * 16 + mloc;
    for (int k0 = 0; k0 < 64; k0 += 4) {
        v2f ar, ai, br, bi;
        #pragma unroll
        for (int r = 0; r < 2; ++r) {
            int j = k0 + r + g * 2;            // f32 A 16x4 / B 4x16 layout
            ar[r] = Kr[j * 64 + mi];
            ai[r] = Ki[j * 64 + mi];
            br[r] = Kr[j * 64 + nk];
            bi[r] = Ki[j * 64 + nk];
        }
        accRR = __builtin_amdgcn_wmma_f32_16x16x4_f32(false, ar, false, br, (short)0, accRR, false, false);
        accII = __builtin_amdgcn_wmma_f32_16x16x4_f32(false, ai, false, bi, (short)0, accII, false, false);
        accRI = __builtin_amdgcn_wmma_f32_16x16x4_f32(false, ar, false, bi, (short)0, accRI, false, false);
        accIR = __builtin_amdgcn_wmma_f32_16x16x4_f32(false, ai, false, br, (short)0, accIR, false, false);
    }
#else
    #pragma unroll
    for (int r = 0; r < 8; ++r) {
        int i2 = ti * 16 + r + g * 8;
        int k2 = tk * 16 + mloc;
        float rr = 0, ii = 0, ri = 0, ir = 0;
        for (int j = 0; j < 64; ++j) {
            float kri = Kr[j*64+i2], kii = Ki[j*64+i2];
            float krk = Kr[j*64+k2], kik = Ki[j*64+k2];
            rr += kri * krk; ii += kii * kik; ri += kri * kik; ir += kii * krk;
        }
        accRR[r] = rr; accII[r] = ii; accRI[r] = ri; accIR[r] = ir;
    }
#endif
    #pragma unroll
    for (int r = 0; r < 8; ++r) {
        int i2 = ti * 16 + r + g * 8;
        int k2 = tk * 16 + mloc;
        float re = accRR[r] + accII[r] + (i2 == k2 ? 1.0f : 0.0f);  // EPS = 1
        float im = accRI[r] - accIR[r];
        size_t o = (((size_t)b * 64 + i2) * 64 + k2) * 2;
        RzOut[o] = re; RzOut[o + 1] = im;
    }
}

// ---------------------------------------------------------------------------
// Hermitian 64x64 eigendecomposition per batch: cyclic complex Jacobi.
// One block (256 threads) per batch; A and V resident in LDS (~68 KB of the
// 320 KB/WGP). Writes U = evecs[:, order[:59]] zero-padded to 64 cols.
// ---------------------------------------------------------------------------
__global__ void eigh_jacobi_kernel(const float* __restrict__ RzOut, float2* __restrict__ U)
{
    __shared__ float2 A[64][66];
    __shared__ float2 V[64][66];
    __shared__ float  rc[32];
    __shared__ float2 rs[32];
    __shared__ int    pp[32], pq[32];
    __shared__ int    order[64];

    const int b = blockIdx.x, tid = threadIdx.x;

    for (int e = tid; e < 4096; e += 256) {
        int i = e >> 6, k = e & 63;
        size_t o = (((size_t)b * 64 + i) * 64 + k) * 2;
        A[i][k] = make_float2(RzOut[o], RzOut[o + 1]);
        V[i][k] = make_float2(i == k ? 1.f : 0.f, 0.f);
    }
    __syncthreads();

    for (int sweep = 0; sweep < 12; ++sweep)
    for (int r = 0; r < 63; ++r) {
        if (tid < 32) {   // tournament pairing: 32 disjoint pairs
            int p, q;
            if (tid == 0) { p = 63; q = r; }
            else { p = (r + tid) % 63; q = (r - tid + 63) % 63; }
            pp[tid] = p; pq[tid] = q;
            float app = A[p][p].x, aqq = A[q][q].x;
            float2 apq = A[p][q];
            float r2 = apq.x * apq.x + apq.y * apq.y;
            float c = 1.f; float2 s = make_float2(0.f, 0.f);
            if (r2 > 1e-24f) {
                float ra  = sqrtf(r2);
                float tau = (aqq - app) / (2.f * ra);
                float t   = (tau >= 0.f ? 1.f : -1.f) / (fabsf(tau) + sqrtf(tau * tau + 1.f));
                c = rsqrtf(1.f + t * t);
                float f = (t * c) / ra;
                s = make_float2(f * apq.x, f * apq.y);
            }
            rc[tid] = c; rs[tid] = s;
        }
        __syncthreads();
        {   // column phase: A <- A*G, V <- V*G
            int t = tid >> 3, sub = tid & 7;
            int p = pp[t], q = pq[t];
            float c = rc[t]; float2 s = rs[t]; float2 sc = make_float2(s.x, -s.y);
            #pragma unroll
            for (int jj = 0; jj < 8; ++jj) {
                int i = sub + jj * 8;
                float2 ap = A[i][p], aq = A[i][q];
                A[i][p] = cadd(cscale(c, ap), cmul(sc, aq));
                A[i][q] = csub(cscale(c, aq), cmul(s, ap));
                float2 vp = V[i][p], vq = V[i][q];
                V[i][p] = cadd(cscale(c, vp), cmul(sc, vq));
                V[i][q] = csub(cscale(c, vq), cmul(s, vp));
            }
        }
        __syncthreads();
        {   // row phase: A <- G^H * A
            int t = tid >> 3, sub = tid & 7;
            int p = pp[t], q = pq[t];
            float c = rc[t]; float2 s = rs[t]; float2 sc = make_float2(s.x, -s.y);
            #pragma unroll
            for (int jj = 0; jj < 8; ++jj) {
                int k = sub + jj * 8;
                float2 mp = A[p][k], mq = A[q][k];
                A[p][k] = cadd(cscale(c, mp), cmul(s, mq));
                A[q][k] = csub(cscale(c, mq), cmul(sc, mp));
            }
        }
        __syncthreads();
    }

    if (tid == 0) {       // ascending eigenvalue order (matches eigh)
        int idx[64];
        for (int i = 0; i < 64; ++i) idx[i] = i;
        for (int i = 0; i < 64; ++i) {
            int best = i;
            for (int j = i + 1; j < 64; ++j)
                if (A[idx[j]][idx[j]].x < A[idx[best]][idx[best]].x) best = j;
            int tmp = idx[i]; idx[i] = idx[best]; idx[best] = tmp;
            order[i] = idx[i];
        }
    }
    __syncthreads();
    for (int e = tid; e < 4096; e += 256) {
        int i = e >> 6, j = e & 63;
        float2 v = (j < 59) ? V[i][order[j]] : make_float2(0.f, 0.f);
        U[(size_t)b * 4096 + i * 64 + j] = v;
    }
}

// ---------------------------------------------------------------------------
// F = U U^H (K padded to 64 with zero cols). Same WMMA structure as rz_kernel.
// ---------------------------------------------------------------------------
__global__ void fproj_kernel(const float2* __restrict__ U, float2* __restrict__ F)
{
    const int lane = threadIdx.x & 31;
    const int g    = lane >> 4;
    const int mloc = lane & 15;
    int bid = blockIdx.x;
    const int tk = bid & 3; bid >>= 2;
    const int ti = bid & 3; bid >>= 2;
    const int b  = bid;
    const float2* Ub = U + (size_t)b * 4096;   // [i][j], stride 64

    v8f accRR = {}, accII = {}, accRI = {}, accIR = {};
#if HAS_WMMA_F32X4
    const int mi = ti * 16 + mloc;
    const int nk = tk * 16 + mloc;
    for (int k0 = 0; k0 < 64; k0 += 4) {
        v2f ar, ai, br, bi;
        #pragma unroll
        for (int r = 0; r < 2; ++r) {
            int j = k0 + r + g * 2;
            float2 ua = Ub[mi * 64 + j];
            float2 ub = Ub[nk * 64 + j];
            ar[r] = ua.x; ai[r] = ua.y;
            br[r] = ub.x; bi[r] = ub.y;
        }
        accRR = __builtin_amdgcn_wmma_f32_16x16x4_f32(false, ar, false, br, (short)0, accRR, false, false);
        accII = __builtin_amdgcn_wmma_f32_16x16x4_f32(false, ai, false, bi, (short)0, accII, false, false);
        accRI = __builtin_amdgcn_wmma_f32_16x16x4_f32(false, ar, false, bi, (short)0, accRI, false, false);
        accIR = __builtin_amdgcn_wmma_f32_16x16x4_f32(false, ai, false, br, (short)0, accIR, false, false);
    }
#else
    #pragma unroll
    for (int r = 0; r < 8; ++r) {
        int i2 = ti * 16 + r + g * 8;
        int k2 = tk * 16 + mloc;
        float rr = 0, ii = 0, ri = 0, ir = 0;
        for (int j = 0; j < 64; ++j) {
            float2 ua = Ub[i2 * 64 + j], ub = Ub[k2 * 64 + j];
            rr += ua.x * ub.x; ii += ua.y * ub.y; ri += ua.x * ub.y; ir += ua.y * ub.x;
        }
        accRR[r] = rr; accII[r] = ii; accRI[r] = ri; accIR[r] = ir;
    }
#endif
    #pragma unroll
    for (int r = 0; r < 8; ++r) {
        int i2 = ti * 16 + r + g * 8;
        int k2 = tk * 16 + mloc;
        // F[i][k] = sum_j U[i][j] * conj(U[k][j])
        F[(size_t)b * 4096 + i2 * 64 + k2] =
            make_float2(accRR[r] + accII[r], accIR[r] - accRI[r]);
    }
}

// ---------------------------------------------------------------------------
// coeffs[d] = trace of (d-63)-th diagonal of F, d in [0, 126].
// ---------------------------------------------------------------------------
__global__ void coeffs_kernel(const float2* __restrict__ F, float2* __restrict__ C)
{
    const int b = blockIdx.x, d = threadIdx.x;
    if (d < 127) {
        int off = d - 63;
        float2 s = make_float2(0.f, 0.f);
        for (int i = 0; i < 64; ++i) {
            int k = i + off;
            if (k >= 0 && k < 64) s = cadd(s, F[(size_t)b * 4096 + i * 64 + k]);
        }
        C[(size_t)b * 127 + d] = s;
    }
}

// ---------------------------------------------------------------------------
// Roots of p(z) = sum_{j=0}^{126} (c_j/c_0) z^{126-j} via Aberth-Ehrlich.
// One block per batch, one lane per root, fp64, fixed 48 iterations.
// ---------------------------------------------------------------------------
__global__ void aberth_kernel(const float2* __restrict__ C, float2* __restrict__ roots,
                              float* __restrict__ rootsLastOut)
{
    __shared__ double2 a[127];
    __shared__ double2 zA[126], zB[126];
    const int b = blockIdx.x, tid = threadIdx.x;

    if (tid < 127) {
        float2 c0 = C[(size_t)b * 127 + 0];
        float2 cj = C[(size_t)b * 127 + tid];
        a[tid] = ddiv(make_double2((double)cj.x, (double)cj.y),
                      make_double2((double)c0.x, (double)c0.y));
    }
    if (tid < 126) {   // MUSIC roots cluster near |z|=1: init near unit circle
        double th  = 6.283185307179586 * ((double)tid / 126.0) + 0.37;
        double rad = 0.9 + 0.2 * ((double)tid / 126.0);
        zA[tid] = make_double2(rad * cos(th), rad * sin(th));
    }
    __syncthreads();

    for (int it = 0; it < 48; ++it) {
        double2* zin  = (it & 1) ? zB : zA;
        double2* zout = (it & 1) ? zA : zB;
        if (tid < 126) {
            double2 z = zin[tid];
            double2 p = a[0], dp = make_double2(0.0, 0.0);
            for (int j = 1; j <= 126; ++j) {           // Horner: p and p'
                dp = dadd(dmul(dp, z), p);
                p  = dadd(dmul(p, z), a[j]);
            }
            double2 N = ddiv(p, dp);
            double2 S = make_double2(0.0, 0.0);
            for (int j = 0; j < 126; ++j) {
                if (j == tid) continue;
                S = dadd(S, ddiv(make_double2(1.0, 0.0), dsub(z, zin[j])));
            }
            double2 den = dsub(make_double2(1.0, 0.0), dmul(N, S));
            zout[tid] = dsub(z, ddiv(N, den));
        }
        __syncthreads();
    }
    if (tid < 126) {   // 48 iterations (even) -> final values in zA
        double2 z = zA[tid];
        roots[(size_t)b * 126 + tid] = make_float2((float)z.x, (float)z.y);
        if (b == 127) {
            rootsLastOut[tid * 2]     = (float)z.x;
            rootsLastOut[tid * 2 + 1] = (float)z.y;
        }
    }
}

// ---------------------------------------------------------------------------
// doa_all, closeness-based top-5 selection, doa_pred.
// ---------------------------------------------------------------------------
__global__ void finalize_kernel(const float2* __restrict__ roots,
                                float* __restrict__ doa_pred, float* __restrict__ doa_all)
{
    __shared__ float close[126];
    __shared__ float ang[126];
    const int b = blockIdx.x, tid = threadIdx.x;
    const float INV_PI = 0.31830988618379067f;

    if (tid < 126) {
        float2 z = roots[(size_t)b * 126 + tid];
        float an = atan2f(z.y, z.x);
        ang[tid] = an;
        doa_all[(size_t)b * 126 + tid] = asinf(an * INV_PI);
        float ab = sqrtf(z.x * z.x + z.y * z.y);
        // inf -> huge finite + index bias to reproduce stable argsort ordering
        close[tid] = (ab < 1.f) ? fabsf(ab - 1.f) : (1e30f + (float)tid);
    }
    __syncthreads();
    if (tid == 0) {
        for (int m = 0; m < 5; ++m) {
            int best = 0; float bv = close[0];
            for (int j = 1; j < 126; ++j)
                if (close[j] < bv) { bv = close[j]; best = j; }
            doa_pred[(size_t)b * 5 + m] = asinf(ang[best] * INV_PI);
            close[best] = 3.0e38f;
        }
    }
}

// ---------------------------------------------------------------------------
// Host launcher
// ---------------------------------------------------------------------------
extern "C" void kernel_launch(void* const* d_in, const int* in_sizes, int n_in,
                              void* d_out, int out_size, void* d_ws, size_t ws_size,
                              hipStream_t stream) {
    (void)in_sizes; (void)n_in; (void)out_size; (void)ws_size;

    const float* Rx_tau = (const float*)d_in[0];
    const float* w1  = (const float*)d_in[1];  const float* b1  = (const float*)d_in[2];
    const float* w2  = (const float*)d_in[3];  const float* b2  = (const float*)d_in[4];
    const float* w3  = (const float*)d_in[5];  const float* b3  = (const float*)d_in[6];
    const float* wd1 = (const float*)d_in[7];  const float* bd1 = (const float*)d_in[8];
    const float* wd2 = (const float*)d_in[9];  const float* bd2 = (const float*)d_in[10];
    const float* wd3 = (const float*)d_in[11]; const float* bd3 = (const float*)d_in[12];

    float* out = (float*)d_out;
    char*  ws  = (char*)d_ws;

    // d_out layout (floats): doa_pred | doa_all | roots[-1] | Rz
    const size_t OUT_PRED  = 0;                 // 128*5
    const size_t OUT_ALL   = 640;               // 128*126
    const size_t OUT_ROOTS = 640 + 16128;       // 126 complex = 252 floats
    const size_t OUT_RZ    = OUT_ROOTS + 252;   // 128*64*64 complex

    // workspace layout (bytes), 256-aligned; x1<->xd2 and x2<->xd1 reuse
    const size_t offX1 = 0;                                    // f16 128*16*127*63
    const size_t offX2 = offX1 + (size_t)16386048 * 2;         // f16 128*32*126*62
    const size_t offX3 = offX2 + (size_t)31997952 * 2;         // f16 128*64*125*61
    const size_t offRX = offX3 + (size_t)62464000 * 2;         // f32 128*128*64
    const size_t offU  = offRX + (size_t)1048576 * 4;          // float2 128*64*64
    const size_t offF  = offU  + (size_t)524288 * 8;           // float2 128*64*64
    const size_t offC  = offF  + (size_t)524288 * 8;           // float2 128*127
    const size_t offR  = offC  + (size_t)(128 * 127) * 8;      // float2 128*126
    const size_t offP  = offR  + (size_t)(128 * 126) * 8;      // probe sink

    _Float16* x1  = (_Float16*)(ws + offX1);
    _Float16* x2  = (_Float16*)(ws + offX2);
    _Float16* x3  = (_Float16*)(ws + offX3);
    _Float16* xd1 = x2;   // reuse
    _Float16* xd2 = x1;   // reuse
    float*    RxF = (float*)(ws + offRX);
    float2*   U   = (float2*)(ws + offU);
    float2*   F   = (float2*)(ws + offF);
    float2*   C   = (float2*)(ws + offC);
    float2*   R   = (float2*)(ws + offR);
    int*      P   = (int*)(ws + offP);

    const int B = 128;

    tdm_probe_kernel<<<dim3(1), 32, 0, stream>>>(P);

    // conv1: (B,8,128,64) -> (B,16,127,63)      block = 32*ntiles
    conv2x2_wmma<float, _Float16, false, true>
        <<<dim3(B * 127 * 4), 32, 0, stream>>>(Rx_tau, w1, b1, x1, 8, 16, 128, 64, 127, 63);
    // conv2: -> (B,32,126,62)
    conv2x2_wmma<_Float16, _Float16, false, true>
        <<<dim3(B * 126 * 4), 64, 0, stream>>>(x1, w2, b2, x2, 16, 32, 127, 63, 126, 62);
    // conv3: -> (B,64,125,61)
    conv2x2_wmma<_Float16, _Float16, false, true>
        <<<dim3(B * 125 * 4), 128, 0, stream>>>(x2, w3, b3, x3, 32, 64, 126, 62, 125, 61);
    // deconv1: -> (B,32,126,62)
    conv2x2_wmma<_Float16, _Float16, true, true>
        <<<dim3(B * 126 * 4), 64, 0, stream>>>(x3, wd1, bd1, xd1, 64, 32, 125, 61, 126, 62);
    // deconv2: -> (B,16,127,63)
    conv2x2_wmma<_Float16, _Float16, true, true>
        <<<dim3(B * 127 * 4), 32, 0, stream>>>(xd1, wd2, bd2, xd2, 32, 16, 126, 62, 127, 63);
    // deconv3: -> (B,1,128,64), no lrelu, f32 output
    conv2x2_wmma<_Float16, float, true, false>
        <<<dim3(B * 128 * 4), 32, 0, stream>>>(xd2, wd3, bd3, RxF, 16, 1, 127, 63, 128, 64);

    rz_kernel<<<dim3(B * 16), 32, 0, stream>>>(RxF, out + OUT_RZ);
    eigh_jacobi_kernel<<<dim3(B), 256, 0, stream>>>(out + OUT_RZ, U);
    fproj_kernel<<<dim3(B * 16), 32, 0, stream>>>(U, F);
    coeffs_kernel<<<dim3(B), 128, 0, stream>>>(F, C);
    aberth_kernel<<<dim3(B), 128, 0, stream>>>(C, R, out + OUT_ROOTS);
    finalize_kernel<<<dim3(B), 128, 0, stream>>>(R, out + OUT_PRED, out + OUT_ALL);
}
